// AllLayerXSA_85572928405670
// MI455X (gfx1250) — matrix operand order repaired and checked by reading, not compiled, not run on hardware
//
#include <hip/hip_runtime.h>
#include <hip/hip_bf16.h>

// Problem constants (from reference): L=16, B=2, T=2048, D=1024
#define DIM     1024
#define NLAYERS 16
#define NTOK    4096   // B*T

typedef __attribute__((ext_vector_type(16))) __bf16 v16bf;
typedef __attribute__((ext_vector_type(8)))  __bf16 v8bf;
typedef __attribute__((ext_vector_type(8)))  float  v8f;

// ---- fp32 -> (bf16 hi, bf16 lo) bits, error-compensated split, manual RNE ----
__device__ __forceinline__ void split_bits(float f, unsigned short& h, unsigned short& l) {
    unsigned u = __float_as_uint(f);
    h = (unsigned short)((u + 0x7fffu + ((u >> 16) & 1u)) >> 16);
    float lf = f - __uint_as_float(((unsigned)h) << 16);
    unsigned u2 = __float_as_uint(lf);
    l = (unsigned short)((u2 + 0x7fffu + ((u2 >> 16) & 1u)) >> 16);
}

// ---- prep 1: fp32 [R][C] -> bf16 hi/lo [C][R] (transpose), LDS-tiled ----
__global__ __launch_bounds__(256) void transpose_split(const float* __restrict__ in,
                                                       unsigned short* __restrict__ ohi,
                                                       unsigned short* __restrict__ olo,
                                                       int R, int C) {
    __shared__ float tile[32][33];
    const int tx = threadIdx.x & 31, ty = threadIdx.x >> 5;
    const int ctiles = C >> 5;
    const int rt = (blockIdx.x / ctiles) << 5;
    const int ct = (blockIdx.x % ctiles) << 5;
#pragma unroll
    for (int i = 0; i < 4; ++i)
        tile[ty + 8 * i][tx] = in[(size_t)(rt + ty + 8 * i) * C + ct + tx];
    __syncthreads();
#pragma unroll
    for (int i = 0; i < 4; ++i) {
        const int cl = ty + 8 * i;
        unsigned short h, l;
        split_bits(tile[tx][cl], h, l);
        const size_t o = (size_t)(ct + cl) * R + rt + tx;
        ohi[o] = h; olo[o] = l;
    }
}

// ---- prep 2: fp32 -> bf16 hi/lo, same layout (one float4 per thread) ----
__global__ __launch_bounds__(256) void split_plain(const float* __restrict__ in,
                                                   unsigned* __restrict__ ohi,
                                                   unsigned* __restrict__ olo) {
    const int i = blockIdx.x * 256 + threadIdx.x;
    float4 v = ((const float4*)in)[i];
    unsigned short h0, l0, h1, l1, h2, l2, h3, l3;
    split_bits(v.x, h0, l0); split_bits(v.y, h1, l1);
    split_bits(v.z, h2, l2); split_bits(v.w, h3, l3);
    ohi[2 * i]     = (unsigned)h0 | ((unsigned)h1 << 16);
    ohi[2 * i + 1] = (unsigned)h2 | ((unsigned)h3 << 16);
    olo[2 * i]     = (unsigned)l0 | ((unsigned)l1 << 16);
    olo[2 * i + 1] = (unsigned)l2 | ((unsigned)l3 << 16);
}

// ---- fragment loads from pre-split bf16 (all contiguous b128) ----
// A: row-major [M,K]; lane M=lane&15; K runs {kb..kb+7} U {kb+16..kb+23}, kb=8*(lane>=16)
__device__ __forceinline__ v16bf load_a16(const __bf16* __restrict__ A, int K,
                                          int row, int k0, int lane) {
    const __bf16* p = A + (size_t)row * K + k0 + ((lane >> 4) * 8);
    v8bf q0 = *(const v8bf*)p;
    v8bf q1 = *(const v8bf*)(p + 16);
    return __builtin_shufflevector(q0, q1, 0, 1, 2, 3, 4, 5, 6, 7,
                                           8, 9, 10, 11, 12, 13, 14, 15);
}
// B: column-major [N,K]; lane N=lane&15; K runs {kb..kb+7} U {kb+16..kb+23}, kb=8*(lane>=16)
__device__ __forceinline__ v16bf load_b16(const __bf16* __restrict__ B, int K,
                                          int col, int k0, int lane) {
    const __bf16* p = B + (size_t)col * K + k0 + ((lane >> 4) * 8);
    v8bf q0 = *(const v8bf*)p;
    v8bf q1 = *(const v8bf*)(p + 16);
    return __builtin_shufflevector(q0, q1, 0, 1, 2, 3, 4, 5, 6, 7,
                                           8, 9, 10, 11, 12, 13, 14, 15);
}

// ---- GEMM: C[M,N] = A*B from pre-split bf16 hi/lo, 3-product compensation ----
// One wave owns a 32x32 tile; block = 8 waves; grids divide exactly (EXEC all-1s).
__global__ __launch_bounds__(256) void gemm_bf16pre(const __bf16* __restrict__ Ahi,
                                                    const __bf16* __restrict__ Alo,
                                                    const __bf16* __restrict__ Bhi,
                                                    const __bf16* __restrict__ Blo,
                                                    float* __restrict__ C,
                                                    int Mrows, int N, int K) {
    const int lane = threadIdx.x & 31;
    const int wave = threadIdx.x >> 5;
    const int mt   = Mrows >> 5;
    const int tile = blockIdx.x * 8 + wave;
    if (tile >= mt * (N >> 5)) return;   // wave-uniform; never taken
    const int m0 = (tile % mt) * 32;
    const int n0 = (tile / mt) * 32;

    v8f acc[2][2];
#pragma unroll
    for (int i = 0; i < 2; ++i)
#pragma unroll
        for (int j = 0; j < 2; ++j)
#pragma unroll
            for (int r = 0; r < 8; ++r) acc[i][j][r] = 0.0f;

    const int ra = lane & 15;
    for (int k0 = 0; k0 < K; k0 += 32) {
        v16bf ahi[2], alo[2], bhi[2], blo[2];
#pragma unroll
        for (int i = 0; i < 2; ++i) {
            ahi[i] = load_a16(Ahi, K, m0 + 16 * i + ra, k0, lane);
            alo[i] = load_a16(Alo, K, m0 + 16 * i + ra, k0, lane);
            bhi[i] = load_b16(Bhi, K, n0 + 16 * i + ra, k0, lane);
            blo[i] = load_b16(Blo, K, n0 + 16 * i + ra, k0, lane);
        }
#pragma unroll
        for (int i = 0; i < 2; ++i)
#pragma unroll
            for (int j = 0; j < 2; ++j) {
                acc[i][j] = __builtin_amdgcn_wmma_f32_16x16x32_bf16(
                    false, ahi[i], false, bhi[j], (short)0, acc[i][j], false, false);
                acc[i][j] = __builtin_amdgcn_wmma_f32_16x16x32_bf16(
                    false, ahi[i], false, blo[j], (short)0, acc[i][j], false, false);
                acc[i][j] = __builtin_amdgcn_wmma_f32_16x16x32_bf16(
                    false, alo[i], false, bhi[j], (short)0, acc[i][j], false, false);
            }
    }

    // C/D layout: VGPR r <-> row r + 8*(lane>=16); col = lane&15
    const int rb = (lane >> 4) * 8;
#pragma unroll
    for (int i = 0; i < 2; ++i)
#pragma unroll
        for (int j = 0; j < 2; ++j)
#pragma unroll
            for (int r = 0; r < 8; ++r)
                C[(size_t)(m0 + 16 * i + rb + r) * N + (n0 + 16 * j + ra)] = acc[i][j][r];
}

// ---- Fused: scores -> softmax -> scale -> renorm -> weighted sum -> LayerNorm ----
__global__ __launch_bounds__(256) void fused_attn_ln(const float* __restrict__ layers,
                                                     const float* __restrict__ P,
                                                     const float* __restrict__ ca_scales,
                                                     const float* __restrict__ ca_temp,
                                                     const float* __restrict__ ln_w,
                                                     const float* __restrict__ ln_b,
                                                     float* __restrict__ out) {
    const int lane  = threadIdx.x & 31;
    const int wave  = threadIdx.x >> 5;
    const int token = blockIdx.x * 8 + wave;     // < NTOK (grid divides exactly)
    const size_t tbase = (size_t)token * DIM;

    float4 pv[8];
#pragma unroll
    for (int c = 0; c < 8; ++c)
        pv[c] = *(const float4*)(P + tbase + c * 128 + lane * 4);

    float score[NLAYERS];
#pragma unroll
    for (int l = 0; l < NLAYERS; ++l) {
        const float4* x = (const float4*)(layers + ((size_t)l * NTOK + token) * DIM);
        float s = 0.0f;
#pragma unroll
        for (int c = 0; c < 8; ++c) {
            float4 xv = x[c * 32 + lane];
            s += pv[c].x * xv.x + pv[c].y * xv.y + pv[c].z * xv.z + pv[c].w * xv.w;
        }
#pragma unroll
        for (int m = 16; m >= 1; m >>= 1) s += __shfl_xor(s, m, 32);
        score[l] = s;
    }

    const float inv = 1.0f / (32.0f * fabsf(ca_temp[0]));  // sqrt(1024) = 32
    float mx = -3.4e38f;
#pragma unroll
    for (int l = 0; l < NLAYERS; ++l) { score[l] *= inv; mx = fmaxf(mx, score[l]); }
    float se = 0.0f, ts = 0.0f, sc[NLAYERS], w[NLAYERS];
#pragma unroll
    for (int l = 0; l < NLAYERS; ++l) {
        sc[l] = ca_scales[l];
        float e = __expf(score[l] - mx);
        w[l] = e * sc[l];
        se += e; ts += w[l];
    }
    const float denom = 1.0f / (ts + 1e-6f * se);
#pragma unroll
    for (int l = 0; l < NLAYERS; ++l) w[l] = w[l] * sc[l] * denom;

    float4 acc[8];
#pragma unroll
    for (int c = 0; c < 8; ++c) acc[c] = make_float4(0.f, 0.f, 0.f, 0.f);
#pragma unroll
    for (int l = 0; l < NLAYERS; ++l) {
        const float4* x = (const float4*)(layers + ((size_t)l * NTOK + token) * DIM);
        const float wl = w[l];
#pragma unroll
        for (int c = 0; c < 8; ++c) {
            float4 xv = x[c * 32 + lane];
            acc[c].x += wl * xv.x; acc[c].y += wl * xv.y;
            acc[c].z += wl * xv.z; acc[c].w += wl * xv.w;
        }
    }

    float s1 = 0.0f, s2 = 0.0f;
#pragma unroll
    for (int c = 0; c < 8; ++c) {
        s1 += acc[c].x + acc[c].y + acc[c].z + acc[c].w;
        s2 += acc[c].x * acc[c].x + acc[c].y * acc[c].y +
              acc[c].z * acc[c].z + acc[c].w * acc[c].w;
    }
#pragma unroll
    for (int m = 16; m >= 1; m >>= 1) { s1 += __shfl_xor(s1, m, 32); s2 += __shfl_xor(s2, m, 32); }
    const float mean = s1 * (1.0f / DIM);
    const float var  = s2 * (1.0f / DIM) - mean * mean;
    const float rstd = rsqrtf(var + 1e-5f);

#pragma unroll
    for (int c = 0; c < 8; ++c) {
        const int off = c * 128 + lane * 4;
        float4 gw = *(const float4*)(ln_w + off);
        float4 gb = *(const float4*)(ln_b + off);
        float4 o;
        o.x = (acc[c].x - mean) * rstd * gw.x + gb.x;
        o.y = (acc[c].y - mean) * rstd * gw.y + gb.y;
        o.z = (acc[c].z - mean) * rstd * gw.z + gb.z;
        o.w = (acc[c].w - mean) * rstd * gw.w + gb.w;
        *(float4*)(out + tbase + off) = o;
    }
}

extern "C" void kernel_launch(void* const* d_in, const int* in_sizes, int n_in,
                              void* d_out, int out_size, void* d_ws, size_t ws_size,
                              hipStream_t stream) {
    (void)in_sizes; (void)n_in; (void)out_size; (void)ws_size;
    const float* layers    = (const float*)d_in[0];  // [L,B,T,D]
    const float* w_q       = (const float*)d_in[1];  // [D,D]
    const float* w_k       = (const float*)d_in[2];  // [D,D]
    const float* ca_scales = (const float*)d_in[3];  // [L]
    const float* ca_temp   = (const float*)d_in[4];  // [1]
    const float* ln_w      = (const float*)d_in[5];  // [D]
    const float* ln_b      = (const float*)d_in[6];  // [D]
    float*       out       = (float*)d_out;          // [B,T,D]
    const float* Xlast = layers + (size_t)(NLAYERS - 1) * NTOK * DIM;

    // ---- workspace layout (48 MB total) ----
    char* ws = (char*)d_ws;
    const size_t DD_F32 = (size_t)DIM * DIM * 4;      // 4 MB
    const size_t DD_BF  = (size_t)DIM * DIM * 2;      // 2 MB
    const size_t XD_BF  = (size_t)NTOK * DIM * 2;     // 8 MB
    float*          Mmat = (float*)ws;                            ws += DD_F32;  // [D,D] f32
    float*          P    = (float*)ws;                            ws += (size_t)NTOK * DIM * 4;
    unsigned short* A1h  = (unsigned short*)ws;                   ws += DD_BF;   // w_q^T hi
    unsigned short* A1l  = (unsigned short*)ws;                   ws += DD_BF;
    unsigned short* B1h  = (unsigned short*)ws;                   ws += DD_BF;   // w_k^T (col-major B) hi
    unsigned short* B1l  = (unsigned short*)ws;                   ws += DD_BF;
    unsigned short* Mch  = (unsigned short*)ws;                   ws += DD_BF;   // M^T (col-major B) hi
    unsigned short* Mcl  = (unsigned short*)ws;                   ws += DD_BF;
    unsigned short* Xh   = (unsigned short*)ws;                   ws += XD_BF;   // x_last hi
    unsigned short* Xl   = (unsigned short*)ws;                   ws += XD_BF;

    const int TT = (DIM / 32) * (DIM / 32);          // 1024 transpose tiles

    // prep: w_q -> A1 = w_q^T (row-major A);  w_k -> B1 = w_k^T (col-major B)
    transpose_split<<<TT, 256, 0, stream>>>(w_q, A1h, A1l, DIM, DIM);
    transpose_split<<<TT, 256, 0, stream>>>(w_k, B1h, B1l, DIM, DIM);
    // prep: x_last -> bf16 hi/lo row-major   (4M floats -> 1M float4 -> 4096 blocks)
    split_plain<<<(NTOK * DIM / 4) / 256, 256, 0, stream>>>(Xlast, (unsigned*)Xh, (unsigned*)Xl);

    // 1) M = w_q^T * w_k       (1024 tiles -> 128 blocks)
    gemm_bf16pre<<<TT / 8, 256, 0, stream>>>((const __bf16*)A1h, (const __bf16*)A1l,
                                             (const __bf16*)B1h, (const __bf16*)B1l,
                                             Mmat, DIM, DIM, DIM);
    // prep: M -> col-major bf16 B operand
    transpose_split<<<TT, 256, 0, stream>>>(Mmat, Mch, Mcl, DIM, DIM);
    // 2) P = x_last * M        (4096 tiles -> 512 blocks)
    gemm_bf16pre<<<(NTOK / 32) * (DIM / 32) / 8, 256, 0, stream>>>(
        (const __bf16*)Xh, (const __bf16*)Xl,
        (const __bf16*)Mch, (const __bf16*)Mcl, P, NTOK, DIM, DIM);
    // 3) fused attention + LayerNorm (1 wave/token)
    fused_attn_ln<<<NTOK / 8, 256, 0, stream>>>(layers, P, ca_scales, ca_temp,
                                                ln_w, ln_b, out);
}
